// VectorQuantizerEMA_61306363183363
// MI455X (gfx1250) — matrix operand order repaired and checked by reading, not compiled, not run on hardware
//
#include <hip/hip_runtime.h>
#include <hip/hip_bf16.h>

// ---------------------------------------------------------------------------
// VectorQuantizerEMA for MI455X (gfx1250, wave32, WMMA + TDM)
//   inputs          [N=32768, D=512] f32
//   embedding       [K=8192,  D=512] f32
//   ema_cluster_size[K] f32
//   ema_w           [K, D] f32
// outputs (flat, concat): quantized_st [N*D], loss[1], perplexity[1],
//                         new_embedding [K*D], new_cs [K], new_w [K*D]
// ---------------------------------------------------------------------------

#define NROWS 32768
#define KEMB  8192
#define DDIM  512
#define NB    128  // embedding columns per outer chunk
#define KB    32   // wmma K step (bf16)
#define LDSBUF_ELEMS (NB * 64)          // 8192 bf16 elems = 16KB per buffer

typedef __attribute__((ext_vector_type(16))) __bf16 v16bf;
typedef __attribute__((ext_vector_type(8)))  float  v8f;
typedef __attribute__((ext_vector_type(4)))  unsigned int v4u;
typedef __attribute__((ext_vector_type(8)))  int    v8i_;
typedef __attribute__((ext_vector_type(4)))  int    v4i_;

#if __has_builtin(__builtin_amdgcn_tensor_load_to_lds) && \
    __has_builtin(__builtin_amdgcn_s_wait_tensorcnt)
#define HAVE_TDM 1
#else
#define HAVE_TDM 0
#endif

// ---------------------------------------------------------------------------
// Prep: split inputs into bf16 hi/lo (grid-stride)
// ---------------------------------------------------------------------------
__global__ __launch_bounds__(256)
void split_x_kernel(const float* __restrict__ X,
                    __hip_bfloat16* __restrict__ Xhi,
                    __hip_bfloat16* __restrict__ Xlo)
{
    const size_t total = (size_t)NROWS * DDIM;
    for (size_t i = blockIdx.x * 256ull + threadIdx.x; i < total;
         i += (size_t)gridDim.x * 256ull) {
        float v = X[i];
        __hip_bfloat16 h = __float2bfloat16(v);
        Xhi[i] = h;
        Xlo[i] = __float2bfloat16(v - __bfloat162float(h));
    }
}

// ---------------------------------------------------------------------------
// Prep: split embedding into interleaved bf16 hi/lo rows:
//   Esp[k] = [512 x hi | 512 x lo]  (row stride 1024 elems)
// plus 0.5*||e_k||^2. One block per embedding row.
// ---------------------------------------------------------------------------
__global__ __launch_bounds__(256)
void prep_embed_kernel(const float* __restrict__ E,
                       __hip_bfloat16* __restrict__ Esp,
                       float* __restrict__ ehn)
{
    const int k = blockIdx.x;
    const float* row = E + (size_t)k * DDIM;
    __hip_bfloat16* orow = Esp + (size_t)k * (2 * DDIM);
    float ss = 0.f;
    for (int d = threadIdx.x; d < DDIM; d += 256) {
        float v = row[d];
        __hip_bfloat16 h = __float2bfloat16(v);
        orow[d]        = h;
        orow[DDIM + d] = __float2bfloat16(v - __bfloat162float(h));
        ss += v * v;
    }
    __shared__ float sred[256];
    sred[threadIdx.x] = ss;
    __syncthreads();
    for (int s = 128; s > 0; s >>= 1) {
        if (threadIdx.x < s) sred[threadIdx.x] += sred[threadIdx.x + s];
        __syncthreads();
    }
    if (threadIdx.x == 0) ehn[k] = 0.5f * sred[0];
}

// ---------------------------------------------------------------------------
// Fragment loaders matching the documented gfx1250 bf16 WMMA VGPR layouts.
// A (16x32): lane m<16 holds row m, K = {0..7, 16..23} (rel. kb);
//            lane m+16 holds row m, K = {8..15, 24..31}.
// B (32x16): lane n<16 holds col n, K = {0..7, 16..23};
//            lane n+16 holds col n, K = {8..15, 24..31}.
// ---------------------------------------------------------------------------
static __device__ __forceinline__ v16bf load_a_frag(const __hip_bfloat16* p)
{
    union { uint4 u[2]; v16bf v; } t;
    t.u[0] = *reinterpret_cast<const uint4*>(p);        // K run +0..7
    t.u[1] = *reinterpret_cast<const uint4*>(p + 16);   // K run +16..23
    return t.v;
}
static __device__ __forceinline__ v16bf load_b_frag(const __hip_bfloat16* p)
{
    union { uint4 u[2]; v16bf v; } t;
    t.u[0] = *reinterpret_cast<const uint4*>(p);        // K +0..7
    t.u[1] = *reinterpret_cast<const uint4*>(p + 16);   // K +16..23
    return t.v;
}

// ---------------------------------------------------------------------------
// TDM: one 3-D descriptor fetches both hi and lo halves of a 128-row x
// 32-col slice of the interleaved Esp tensor into LDS at byte offset lds_off.
//   x: within-plane position, extent 512, stride0 = 512 (plane step)
//   y: plane (hi/lo),         extent 2,   stride1 = 1024 (row step)
//   z: embedding row,         extent 8192
//   tile = 32 x 2 x 128  ->  LDS rows of [32 hi | 32 lo] (64 elems/row)
// D# fields per CDNA5 ISA.
// ---------------------------------------------------------------------------
#if HAVE_TDM
static __device__ __forceinline__ void tdm_load_tile3d(unsigned lds_off,
                                                       const __hip_bfloat16* gptr)
{
    const unsigned long long ga = (unsigned long long)(size_t)gptr;
    v4u g0;
    g0.x = 1u;                                            // count=1, user mode
    g0.y = lds_off;                                       // lds_addr (bytes)
    g0.z = (unsigned)(ga & 0xFFFFFFFFu);                  // global_addr[31:0]
    g0.w = (unsigned)((ga >> 32) & 0x01FFFFFFu)           // global_addr[56:32]
         | (2u << 30);                                    // type = 2 (image)
    v8i_ g1;
    g1[0] = 0x00010000;                 // wg_mask=0, data_size=1 (2 bytes)
    g1[1] = (int)(512u << 16);          // tensor_dim0 = 512  (bits 79:48)
    g1[2] = (int)(2u << 16);            // tensor_dim1 = 2    (bits 111:80)
    g1[3] = (int)(32u << 16);           // tile_dim0 = 32     (bits 127:112)
    g1[4] = (int)(2u | ((unsigned)NB << 16));  // tile_dim1 = 2, tile_dim2 = NB
    g1[5] = 512;                        // tensor_dim0_stride[31:0] = 512
    g1[6] = (int)(1024u << 16);         // stride0 hi=0 | tensor_dim1_stride lo16
    g1[7] = 0;                          // tensor_dim1_stride[47:16] = 0
    v4i_ g2;
    g2[0] = KEMB;                       // tensor_dim2 = 8192
    g2[1] = 0;                          // tensor_dim3 = 0
    g2[2] = 0;                          // tensor_dim2_stride (unused, no dim3)
    g2[3] = 0;                          // tile_dim3 = 0
    const v4i_ z4 = {0, 0, 0, 0};       // group 3 unused
    const v8i_ z8 = {0, 0, 0, 0, 0, 0, 0, 0};
    __builtin_amdgcn_tensor_load_to_lds(g0, g1, g2, z4, z8, 0);
}
#endif

// ---------------------------------------------------------------------------
// Heavy kernel: fused S = X.E^T (bf16x3 via v_wmma_f32_16x16x32_bf16)
// with running argmax of (S - 0.5*||e||^2)  ==  argmin of L2 distance.
// 256 blocks x 256 threads (8 waves). Wave w owns 16 rows. E tiles are
// staged into LDS (double-buffered) by the Tensor Data Mover (wave 0),
// synchronized with s_wait_tensorcnt + workgroup barrier.
// LDS (32KB dynamic): buffer b at b*16384 bytes; NB rows x [32 hi | 32 lo].
// Per stage: 24 WMMAs, 32 ds_load_b128, 1 TDM issue, 1 barrier.
// ---------------------------------------------------------------------------
__global__ __launch_bounds__(256)
void argmin_gemm_kernel(const __hip_bfloat16* __restrict__ Xhi,
                        const __hip_bfloat16* __restrict__ Xlo,
                        const __hip_bfloat16* __restrict__ Esp,
                        const float* __restrict__ ehn,
                        int* __restrict__ idx_out)
{
    extern __shared__ char smem[];
    __hip_bfloat16* lds = (__hip_bfloat16*)smem;

    const int wave = threadIdx.x >> 5;          // 0..7
    const int lane = threadIdx.x & 31;
    const int half = lane >> 4;                 // 0 or 1
    const int l16  = lane & 15;
    const int m0   = blockIdx.x * 128 + wave * 16;

    // A row pointers with the per-half K shift folded in (+8 for lanes>=16)
    const __hip_bfloat16* xhi_row = Xhi + (size_t)(m0 + l16) * DDIM + half * 8;
    const __hip_bfloat16* xlo_row = Xlo + (size_t)(m0 + l16) * DDIM + half * 8;

    float best[8];
    int   bidx[8];
#pragma unroll
    for (int r = 0; r < 8; ++r) { best[r] = -3.0e38f; bidx[r] = 0; }

#if !HAVE_TDM
    // fallback cooperative copy: thread -> (row, hi/lo half), 32 elems each
    const int frow = threadIdx.x >> 1;                    // 0..127
    const int fseg = threadIdx.x & 1;                     // 0: hi, 1: lo
#endif

    // ---- stage 0 prologue: tile (nb=0, kb=0) into buffer 0 ----
#if HAVE_TDM
    if (wave == 0) tdm_load_tile3d(0u, Esp);
    __builtin_amdgcn_s_wait_tensorcnt(0);
#else
    {
        const __hip_bfloat16* src = Esp + (size_t)frow * (2 * DDIM) + fseg * DDIM;
        __hip_bfloat16* dst = lds + frow * 64 + fseg * 32;
#pragma unroll
        for (int q = 0; q < 4; ++q)
            *reinterpret_cast<uint4*>(dst + q * 8) =
                *reinterpret_cast<const uint4*>(src + q * 8);
    }
#endif
    __syncthreads();

    int buf = 0;
    for (int nbi = 0; nbi < KEMB / NB; ++nbi) {
        v8f acc[NB / 16];
#pragma unroll
        for (int t = 0; t < NB / 16; ++t)
            acc[t] = (v8f){0.f, 0.f, 0.f, 0.f, 0.f, 0.f, 0.f, 0.f};

        for (int kbi = 0; kbi < DDIM / KB; ++kbi) {
            const int kb = kbi * KB;

            // issue next stage's tile into the other buffer (it was last
            // read in the previous stage; the previous barrier protects it)
            const bool last = (nbi == KEMB / NB - 1) && (kbi == DDIM / KB - 1);
            if (!last) {
                const int kbn = (kbi == DDIM / KB - 1) ? 0 : (kbi + 1);
                const int nbn = (kbi == DDIM / KB - 1) ? (nbi + 1) : nbi;
                const size_t goff = (size_t)(nbn * NB) * (2 * DDIM) + kbn * KB;
#if HAVE_TDM
                if (wave == 0)
                    tdm_load_tile3d((unsigned)((buf ^ 1) * (LDSBUF_ELEMS * 2)),
                                    Esp + goff);
#else
                __syncthreads();        // done reading the other buffer
                const __hip_bfloat16* src =
                    Esp + goff + (size_t)frow * (2 * DDIM) + fseg * DDIM;
                __hip_bfloat16* dst =
                    lds + (buf ^ 1) * LDSBUF_ELEMS + frow * 64 + fseg * 32;
#pragma unroll
                for (int q = 0; q < 4; ++q)
                    *reinterpret_cast<uint4*>(dst + q * 8) =
                        *reinterpret_cast<const uint4*>(src + q * 8);
#endif
            }

            const v16bf ahi = load_a_frag(xhi_row + kb);
            const v16bf alo = load_a_frag(xlo_row + kb);

            const __hip_bfloat16* bt = lds + buf * LDSBUF_ELEMS;
#pragma unroll
            for (int t = 0; t < NB / 16; ++t) {
                const int roff = (t * 16 + l16) * 64 + half * 8;
                const v16bf bhi = load_b_frag(bt + roff);
                const v16bf blo = load_b_frag(bt + roff + 32);
                acc[t] = __builtin_amdgcn_wmma_f32_16x16x32_bf16(
                    false, ahi, false, bhi, (short)0, acc[t], false, false);
                acc[t] = __builtin_amdgcn_wmma_f32_16x16x32_bf16(
                    false, ahi, false, blo, (short)0, acc[t], false, false);
                acc[t] = __builtin_amdgcn_wmma_f32_16x16x32_bf16(
                    false, alo, false, bhi, (short)0, acc[t], false, false);
            }

#if HAVE_TDM
            __builtin_amdgcn_s_wait_tensorcnt(0);  // wave0 waits; others no-op
#endif
            __syncthreads();
            buf ^= 1;
        }

        // end of this N-chunk: fold in -0.5||e||^2, update running argmax
        const int nb = nbi * NB;
#pragma unroll
        for (int t = 0; t < NB / 16; ++t) {
            const int n = nb + t * 16 + l16;
            const float off = ehn[n];
#pragma unroll
            for (int r = 0; r < 8; ++r) {
                const float tv = acc[t][r] - off;
                if (tv > best[r]) { best[r] = tv; bidx[r] = n; }
            }
        }
    }

    // reduce across the 16 lanes holding the same rows (xor 8,4,2,1 stays
    // inside each half of the wave). Ties -> smaller index (first min).
#pragma unroll
    for (int off = 8; off >= 1; off >>= 1) {
#pragma unroll
        for (int r = 0; r < 8; ++r) {
            const float ov = __shfl_xor(best[r], off, 32);
            const int   oi = __shfl_xor(bidx[r], off, 32);
            if (ov > best[r] || (ov == best[r] && oi < bidx[r])) {
                best[r] = ov; bidx[r] = oi;
            }
        }
    }
    // C layout: VGPR r, lanes 0-15 -> row r; lanes 16-31 -> row r+8
    if (l16 == 0) {
#pragma unroll
        for (int r = 0; r < 8; ++r)
            idx_out[m0 + half * 8 + r] = bidx[r];
    }
}

// ---------------------------------------------------------------------------
// Gather codewords, quantized_st output, loss partial, counts & dw scatter.
// One block per input row.
// ---------------------------------------------------------------------------
__global__ __launch_bounds__(256)
void gather_stats_kernel(const float* __restrict__ X,
                         const float* __restrict__ E,
                         const int* __restrict__ idx,
                         float* __restrict__ out_q,
                         float* __restrict__ dw,      // = out_w region, zeroed
                         float* __restrict__ counts,
                         float* __restrict__ loss_acc)
{
    const int n = blockIdx.x;
    const int k = idx[n];
    const float* xrow = X + (size_t)n * DDIM;
    const float* erow = E + (size_t)k * DDIM;
    float* qrow  = out_q + (size_t)n * DDIM;
    float* dwrow = dw    + (size_t)k * DDIM;

    float lsum = 0.f;
    for (int d = threadIdx.x; d < DDIM; d += 256) {
        const float xv = xrow[d];
        const float ev = erow[d];
        qrow[d] = xv + (ev - xv);           // quantized straight-through value
        const float df = ev - xv;
        lsum += df * df;
        atomicAdd(&dwrow[d], xv);           // dw[k] += x
    }
    __shared__ float sred[256];
    sred[threadIdx.x] = lsum;
    __syncthreads();
    for (int s = 128; s > 0; s >>= 1) {
        if (threadIdx.x < s) sred[threadIdx.x] += sred[threadIdx.x + s];
        __syncthreads();
    }
    if (threadIdx.x == 0) {
        atomicAdd(loss_acc, sred[0]);
        atomicAdd(&counts[k], 1.0f);
    }
}

// ---------------------------------------------------------------------------
// Finalize cluster sizes, loss, perplexity. Single block of 1024 threads.
// ---------------------------------------------------------------------------
__global__ __launch_bounds__(1024)
void finalize_cs_kernel(const float* __restrict__ counts,
                        const float* __restrict__ ema_cs,
                        const float* __restrict__ loss_acc,
                        float* __restrict__ out_loss,
                        float* __restrict__ out_perp,
                        float* __restrict__ out_cs)
{
    __shared__ float s_n[1024];
    __shared__ float s_p[1024];
    const int tid = threadIdx.x;

    float raw[KEMB / 1024];
    float nsum = 0.f, psum = 0.f;
#pragma unroll
    for (int i = 0; i < KEMB / 1024; ++i) {
        const int k = tid + i * 1024;
        const float c = counts[k];
        const float r = ema_cs[k] * 0.99f + 0.01f * c;
        raw[i] = r;
        nsum += r;
        const float p = c / (float)NROWS;
        psum += p * logf(p + 1e-10f);
    }
    s_n[tid] = nsum; s_p[tid] = psum;
    __syncthreads();
    for (int s = 512; s > 0; s >>= 1) {
        if (tid < s) { s_n[tid] += s_n[tid + s]; s_p[tid] += s_p[tid + s]; }
        __syncthreads();
    }
    const float n_total = s_n[0];
    const float denom   = n_total + (float)KEMB * 1e-5f;
#pragma unroll
    for (int i = 0; i < KEMB / 1024; ++i) {
        const int k = tid + i * 1024;
        out_cs[k] = (raw[i] + 1e-5f) / denom * n_total;
    }
    if (tid == 0) {
        out_loss[0] = 0.25f * loss_acc[0] / ((float)NROWS * (float)DDIM);
        out_perp[0] = expf(-s_p[0]);
    }
}

// ---------------------------------------------------------------------------
// new_w = ema_w*0.99 + 0.01*dw (in place over the dw region);
// new_embedding = new_w / new_cs[k]
// ---------------------------------------------------------------------------
__global__ __launch_bounds__(256)
void finalize_w_kernel(const float* __restrict__ ema_w,
                       const float* __restrict__ new_cs,
                       float* __restrict__ out_w,    // holds dw on entry
                       float* __restrict__ out_emb)
{
    const size_t total = (size_t)KEMB * DDIM;
    for (size_t i = blockIdx.x * 256ull + threadIdx.x; i < total;
         i += (size_t)gridDim.x * 256ull) {
        const float dwv = out_w[i];
        const float nw  = ema_w[i] * 0.99f + 0.01f * dwv;
        out_w[i]   = nw;
        out_emb[i] = nw / new_cs[i >> 9];   // i / DDIM
    }
}

// ---------------------------------------------------------------------------
extern "C" void kernel_launch(void* const* d_in, const int* in_sizes, int n_in,
                              void* d_out, int out_size, void* d_ws, size_t ws_size,
                              hipStream_t stream)
{
    const float* X      = (const float*)d_in[0];
    const float* E      = (const float*)d_in[1];
    const float* ema_cs = (const float*)d_in[2];
    const float* ema_w  = (const float*)d_in[3];

    float* out      = (float*)d_out;
    float* out_q    = out;                                   // N*D
    float* out_loss = out + (size_t)NROWS * DDIM;            // 1
    float* out_perp = out_loss + 1;                          // 1
    float* out_emb  = out_perp + 1;                          // K*D
    float* out_cs   = out_emb + (size_t)KEMB * DDIM;         // K
    float* out_w    = out_cs + KEMB;                         // K*D (dw accum)

    // workspace layout (16B aligned sections)
    char* ws = (char*)d_ws;
    __hip_bfloat16* Xhi = (__hip_bfloat16*)ws; ws += (size_t)NROWS * DDIM * 2;
    __hip_bfloat16* Xlo = (__hip_bfloat16*)ws; ws += (size_t)NROWS * DDIM * 2;
    __hip_bfloat16* Esp = (__hip_bfloat16*)ws; ws += (size_t)KEMB * DDIM * 4;
    float* ehn      = (float*)ws;  ws += (size_t)KEMB * 4;
    float* counts   = (float*)ws;  ws += (size_t)KEMB * 4;
    float* loss_acc = (float*)ws;  ws += 16;
    int*   idx      = (int*)ws;    ws += (size_t)NROWS * 4;

    // zero accumulators (dw region in d_out, counts+loss in ws)
    (void)hipMemsetAsync(out_w, 0, (size_t)KEMB * DDIM * sizeof(float), stream);
    (void)hipMemsetAsync(counts, 0, (size_t)KEMB * sizeof(float) + 16, stream);

    split_x_kernel<<<8192, 256, 0, stream>>>(X, Xhi, Xlo);
    prep_embed_kernel<<<KEMB, 256, 0, stream>>>(E, Esp, ehn);
    argmin_gemm_kernel<<<NROWS / 128, 256, 32768, stream>>>(Xhi, Xlo, Esp,
                                                            ehn, idx);
    gather_stats_kernel<<<NROWS, 256, 0, stream>>>(X, E, idx, out_q, out_w,
                                                   counts, loss_acc);
    finalize_cs_kernel<<<1, 1024, 0, stream>>>(counts, ema_cs, loss_acc,
                                               out_loss, out_perp, out_cs);
    finalize_w_kernel<<<4096, 256, 0, stream>>>(ema_w, out_cs, out_w, out_emb);
}
